// Attention_74534862455272
// MI455X (gfx1250) — matrix operand order repaired and checked by reading, not compiled, not run on hardware
//
#include <hip/hip_runtime.h>

typedef __attribute__((ext_vector_type(16))) _Float16 v16h;
typedef __attribute__((ext_vector_type(8)))  _Float16 v8h;
typedef __attribute__((ext_vector_type(8)))  float    v8f;
typedef __attribute__((ext_vector_type(4)))  int      v4i;

#define B_  2
#define T_  2048
#define C_  1024
#define H_  16
#define D_  64
#define DH_ 32   // D/2 (RoPE half)

#if defined(__has_builtin)
#if __has_builtin(__builtin_amdgcn_global_load_async_to_lds_b128)
#define HAVE_ASYNC_LDS 1
#endif
#endif

static __device__ __forceinline__ v8f wmma_f16(v16h a, v16h b, v8f c) {
  // D = A(16x32 f16) * B(32x16 f16) + C(16x16 f32)
  return __builtin_amdgcn_wmma_f32_16x16x32_f16(false, a, false, b, (short)0, c,
                                                false, false);
}

static __device__ __forceinline__ v16h pack16(v8h lo, v8h hi) {
  v16h r;
#pragma unroll
  for (int j = 0; j < 8; ++j) { r[j] = lo[j]; r[j + 8] = hi[j]; }
  return r;
}

// ---------------------------------------------------------------------------
// Kernel 0: one-shot f32 -> f16 conversion (x, W_qkv, W_out)
// ---------------------------------------------------------------------------
__global__ void cvt_f16_kernel(const float* __restrict__ src,
                               _Float16* __restrict__ dst, int n) {
  const int i = (blockIdx.x * blockDim.x + threadIdx.x) * 8;
  if (i < n) {
#pragma unroll
    for (int j = 0; j < 8; ++j) dst[i + j] = (_Float16)src[i + j];
  }
}

// ---------------------------------------------------------------------------
// Kernel 1: qkv = x @ W_qkv^T  (16x64 tile per wave; 8 waves share a block-
// staged 16x1024 f16 A tile in LDS), RoPE on q/k.
//   q,k -> [B,H,T,D] f16 ; v -> [B,H,D,T] (transposed) f16
// ---------------------------------------------------------------------------
__global__ void qkv_rope_kernel(const _Float16* __restrict__ x16,
                                const float* __restrict__ cosb,
                                const float* __restrict__ sinb,
                                const _Float16* __restrict__ wq16,
                                _Float16* __restrict__ qws,
                                _Float16* __restrict__ kws,
                                _Float16* __restrict__ vtws) {
  __shared__ _Float16 atile[16 * C_];              // 32 KB

  const int lane = threadIdx.x & 31;
  const int wib  = threadIdx.x >> 5;               // 0..7
  const int n16  = lane & 15;
  const int hi   = lane >> 4;

  const int rowTile   = blockIdx.x / 6;            // 0..255
  const int colGroup  = blockIdx.x % 6;            // 0..5
  const int mbase     = rowTile * 16;
  const int nbase     = (colGroup * 8 + wib) * 64; // head-sized column block

  // ---- cooperative stage of A tile (16 rows x 1024 cols f16 = 32 KB) ----
  {
    const _Float16* src = x16 + (size_t)mbase * C_;
#if defined(HAVE_ASYNC_LDS)
#pragma unroll
    for (int c = 0; c < 8; ++c) {
      const int elt = (c * 256 + (int)threadIdx.x) * 8;   // 8 halves = 16 B
      __builtin_amdgcn_global_load_async_to_lds_b128(
          (v4i*)(src + elt), (v4i*)(atile + elt), 0, 0);
    }
    asm volatile("s_wait_asynccnt 0" ::: "memory");
#else
#pragma unroll
    for (int c = 0; c < 8; ++c) {
      const int elt = (c * 256 + (int)threadIdx.x) * 8;
      *(v8h*)(atile + elt) = *(const v8h*)(src + elt);
    }
#endif
  }
  __syncthreads();

  const int arow  = n16;            // A row (within tile) for this lane
  const int abase = hi * 16;        // A-layout K contiguous base

  v8f acc[4] = {};
  for (int kc = 0; kc < C_; kc += 32) {
    const _Float16* xp = atile + arow * C_ + kc;
    v16h a = pack16(*(const v8h*)(xp + abase), *(const v8h*)(xp + abase + 8));
#pragma unroll
    for (int nt = 0; nt < 4; ++nt) {
      const _Float16* wp =
          wq16 + (size_t)(nbase + nt * 16 + n16) * C_ + kc + hi * 16;
      v16h b = *(const v16h*)wp;
      acc[nt] = wmma_f16(a, b, acc[nt]);
    }
  }

  const int which = nbase / C_;          // 0=q 1=k 2=v
  const int h     = (nbase % C_) / D_;   // head index (block == one head)

  if (which == 2) {
    // V: store transposed [B,H,D,T]
#pragma unroll
    for (int nt = 0; nt < 4; ++nt) {
#pragma unroll
      for (int r = 0; r < 8; ++r) {
        const int m    = mbase + r + 8 * hi;
        const int bb   = m / T_;
        const int tpos = m % T_;
        const int dd   = nt * 16 + n16;
        vtws[((size_t)(bb * H_ + h) * D_ + dd) * T_ + tpos] = (_Float16)acc[nt][r];
      }
    }
  } else {
    _Float16* dst = (which == 0) ? qws : kws;
#pragma unroll
    for (int r = 0; r < 8; ++r) {
      const int m    = mbase + r + 8 * hi;
      const int bb   = m / T_;
      const int tpos = m % T_;
      _Float16* drow = dst + ((size_t)(bb * H_ + h) * T_ + tpos) * D_;
#pragma unroll
      for (int t = 0; t < 2; ++t) {
        const int dd = t * 16 + n16;   // < 32
        const float c  = cosb[tpos * DH_ + dd];
        const float s  = sinb[tpos * DH_ + dd];
        const float x1 = acc[t][r];        // dims [0,32)
        const float x2 = acc[t + 2][r];    // dims [32,64)
        drow[dd]       = (_Float16)(x1 * c - x2 * s);
        drow[dd + 32]  = (_Float16)(x2 * c + x1 * s);
      }
    }
  }
}

// ---------------------------------------------------------------------------
// Kernel 2: causal flash attention, one wave per (b,h, 16-query-row tile).
//   S = Q K^T (WMMA), online softmax, O += P V (WMMA via per-wave LDS restage)
// ---------------------------------------------------------------------------
__global__ void flash_attn_kernel(const _Float16* __restrict__ qws,
                                  const _Float16* __restrict__ kws,
                                  const _Float16* __restrict__ vtws,
                                  _Float16* __restrict__ attn) {
  __shared__ _Float16 plds[4][16 * 32];   // per-wave P staging tile

  const int wave = (blockIdx.x * blockDim.x + threadIdx.x) >> 5;
  const int wib  = (threadIdx.x >> 5) & 3;
  const int lane = threadIdx.x & 31;
  const int n16  = lane & 15;
  const int hi   = lane >> 4;

  const int qTiles = T_ / 16;              // 128
  const int bh     = wave / qTiles;        // 0..31
  const int qbase  = (wave % qTiles) * 16;

  const _Float16* qp  = qws  + (size_t)bh * T_ * D_;
  const _Float16* kp  = kws  + (size_t)bh * T_ * D_;
  const _Float16* vtp = vtws + (size_t)bh * D_ * T_;

  // Preload Q A-fragments (two K=32 chunks covering D=64)
  v16h aq[2];
  {
    const int row  = qbase + n16;
    const int base = hi * 16;
#pragma unroll
    for (int c = 0; c < 2; ++c) {
      const _Float16* p = qp + (size_t)row * D_ + c * 32;
      aq[c] = pack16(*(const v8h*)(p + base), *(const v8h*)(p + base + 8));
    }
  }

  v8f o[4] = {};
  float m_i[8], l_i[8];
#pragma unroll
  for (int r = 0; r < 8; ++r) { m_i[r] = -__builtin_inff(); l_i[r] = 0.f; }

  for (int kb = 0; kb < qbase + 16; kb += 32) {
    // ---- S = Q K^T for 32 keys (two 16-wide tiles) ----
    v8f s[2] = {};
#pragma unroll
    for (int t = 0; t < 2; ++t) {
      const int key = kb + t * 16 + n16;
#pragma unroll
      for (int c = 0; c < 2; ++c) {
        v16h bk = *(const v16h*)(kp + (size_t)key * D_ + c * 32 + hi * 16);
        s[t] = wmma_f16(aq[c], bk, s[t]);
      }
    }

    // ---- scale, causal mask, online softmax ----
    float p[2][8];
#pragma unroll
    for (int r = 0; r < 8; ++r) {
      const int qpos = qbase + r + 8 * hi;
#pragma unroll
      for (int t = 0; t < 2; ++t) {
        const int kpos = kb + t * 16 + n16;
        float v = s[t][r] * 0.125f;                    // 1/sqrt(64)
        if (kpos > qpos) v = -__builtin_inff();
        s[t][r] = v;
      }
      float v = fmaxf(s[0][r], s[1][r]);
#pragma unroll
      for (int msk = 1; msk < 16; msk <<= 1) v = fmaxf(v, __shfl_xor(v, msk, 32));
      const float mnew  = fmaxf(m_i[r], v);
      const float alpha = __expf(m_i[r] - mnew);
      float rs = 0.f;
#pragma unroll
      for (int t = 0; t < 2; ++t) {
        const float e = __expf(s[t][r] - mnew);
        p[t][r] = e;
        rs += e;
      }
#pragma unroll
      for (int msk = 1; msk < 16; msk <<= 1) rs += __shfl_xor(rs, msk, 32);
      l_i[r] = l_i[r] * alpha + rs;
      m_i[r] = mnew;
#pragma unroll
      for (int nt = 0; nt < 4; ++nt) o[nt][r] *= alpha;
    }

    // ---- restage P (C/D layout -> A layout) through per-wave LDS ----
    _Float16* pl = &plds[wib][0];
#pragma unroll
    for (int t = 0; t < 2; ++t)
#pragma unroll
      for (int r = 0; r < 8; ++r)
        pl[(r + 8 * hi) * 32 + t * 16 + n16] = (_Float16)p[t][r];
    asm volatile("s_wait_dscnt 0" ::: "memory");
    const _Float16* pr = pl + n16 * 32;
    v16h ap = pack16(*(const v8h*)(pr + hi * 16), *(const v8h*)(pr + hi * 16 + 8));

    // ---- O += P V  (V transposed => contiguous B fragments) ----
#pragma unroll
    for (int nt = 0; nt < 4; ++nt) {
      v16h bv = *(const v16h*)(vtp + (size_t)(nt * 16 + n16) * T_ + kb + hi * 16);
      o[nt] = wmma_f16(ap, bv, o[nt]);
    }
  }

  // ---- normalize and write attention output as f16 [B,T,C] ----
  const int bb = bh / H_;
  const int h  = bh % H_;
#pragma unroll
  for (int r = 0; r < 8; ++r) {
    const int qpos  = qbase + r + 8 * hi;
    const float inv = 1.f / l_i[r];
    _Float16* arow = attn + (size_t)(bb * T_ + qpos) * C_ + h * D_;
#pragma unroll
    for (int nt = 0; nt < 4; ++nt)
      arow[nt * 16 + n16] = (_Float16)(o[nt][r] * inv);
  }
}

// ---------------------------------------------------------------------------
// Kernel 3: out = attn @ W_out^T  (16x64 tile per wave), f32 output
// ---------------------------------------------------------------------------
__global__ void out_proj_kernel(const _Float16* __restrict__ attn,
                                const _Float16* __restrict__ wo16,
                                float* __restrict__ out) {
  const int wave = (blockIdx.x * blockDim.x + threadIdx.x) >> 5;
  const int lane = threadIdx.x & 31;
  const int n16  = lane & 15;
  const int hi   = lane >> 4;

  const int nColBlocks = C_ / 64;                 // 16
  const int mbase = (wave / nColBlocks) * 16;
  const int nbase = (wave % nColBlocks) * 64;

  const int arow  = mbase + n16;
  const int abase = hi * 16;

  v8f acc[4] = {};
  for (int kc = 0; kc < C_; kc += 32) {
    const _Float16* ax = attn + (size_t)arow * C_ + kc;
    v16h a = pack16(*(const v8h*)(ax + abase), *(const v8h*)(ax + abase + 8));
#pragma unroll
    for (int nt = 0; nt < 4; ++nt) {
      const _Float16* wp =
          wo16 + (size_t)(nbase + nt * 16 + n16) * C_ + kc + hi * 16;
      v16h b = *(const v16h*)wp;
      acc[nt] = wmma_f16(a, b, acc[nt]);
    }
  }
#pragma unroll
  for (int nt = 0; nt < 4; ++nt)
#pragma unroll
    for (int r = 0; r < 8; ++r)
      out[(size_t)(mbase + r + 8 * hi) * C_ + nbase + nt * 16 + n16] = acc[nt][r];
}

// ---------------------------------------------------------------------------
extern "C" void kernel_launch(void* const* d_in, const int* in_sizes, int n_in,
                              void* d_out, int out_size, void* d_ws, size_t ws_size,
                              hipStream_t stream) {
  const float* x    = (const float*)d_in[0];
  const float* cosb = (const float*)d_in[1];
  const float* sinb = (const float*)d_in[2];
  // d_in[3] = mask (unused: causal structure is implicit)
  const float* Wqkv = (const float*)d_in[4];
  const float* Wout = (const float*)d_in[5];
  float* out = (float*)d_out;

  const size_t N_X  = (size_t)B_ * T_ * C_;        // 4,194,304
  const size_t N_WQ = (size_t)3 * C_ * C_;         // 3,145,728
  const size_t N_WO = (size_t)C_ * C_;             // 1,048,576
  const size_t BHTD = (size_t)B_ * H_ * T_ * D_;   // 4,194,304

  _Float16* x16  = (_Float16*)d_ws;                // [B*T, C]
  _Float16* wq16 = x16  + N_X;                     // [3C, C]
  _Float16* wo16 = wq16 + N_WQ;                    // [C, C]
  _Float16* qws  = wo16 + N_WO;                    // [B,H,T,D]
  _Float16* kws  = qws  + BHTD;                    // [B,H,T,D]
  _Float16* vtws = kws  + BHTD;                    // [B,H,D,T]
  _Float16* attn = vtws + BHTD;                    // [B,T,C] f16

  // ---- pass 0: pre-convert everything to f16 (bandwidth-bound, ~1.5 us) ----
  cvt_f16_kernel<<<(int)(N_X  / (8 * 256)), 256, 0, stream>>>(x,    x16,  (int)N_X);
  cvt_f16_kernel<<<(int)(N_WQ / (8 * 256)), 256, 0, stream>>>(Wqkv, wq16, (int)N_WQ);
  cvt_f16_kernel<<<(int)(N_WO / (8 * 256)), 256, 0, stream>>>(Wout, wo16, (int)N_WO);

  // 1536 blocks: 256 row tiles x 6 column groups (8 waves/block)
  qkv_rope_kernel<<<1536, 256, 0, stream>>>(x16, cosb, sinb, wq16, qws, kws, vtws);
  // 4096 waves: 32 (b,h) x 128 query tiles
  flash_attn_kernel<<<1024, 128, 0, stream>>>(qws, kws, vtws, attn);
  // 4096 waves: 256 row tiles x 16 column blocks
  out_proj_kernel<<<512, 256, 0, stream>>>(attn, wo16, out);
}
